// SSIM_64304250356077
// MI455X (gfx1250) — matrix operand order repaired and checked by reading, not compiled, not run on hardware
//
#include <hip/hip_runtime.h>

typedef __attribute__((ext_vector_type(16))) _Float16 v16h;
typedef __attribute__((ext_vector_type(8)))  float    v8f;
typedef __attribute__((ext_vector_type(2)))  _Float16 h2;

#define IMG_H 512
#define IMG_W 960
#define TILE  16
#define PAD   5
#define PATCH 26            // TILE + 2*PAD
#define SSIM_C0 1.0e-4f     // 0.01^2
#define SSIM_C1 9.0e-4f     // 0.03^2

// Normalized 1-D Gaussian, w=11, sigma=1.5 (matches reference _gaussian_kernel_2d)
__constant__ float GK[11] = {
    1.0283791e-03f, 7.5988197e-03f, 3.6000773e-02f, 1.0936069e-01f,
    2.1300553e-01f, 2.6601172e-01f, 2.1300553e-01f, 1.0936069e-01f,
    3.6000773e-02f, 7.5988197e-03f, 1.0283791e-03f};

__global__ __launch_bounds__(32)
void ssim_wmma_kernel(const float* __restrict__ img0,
                      const float* __restrict__ img1,
                      float* __restrict__ out) {
    // fp16 channel patches: rows/cols 26..31 zero-padded so they feed 0 into WMMA
    __shared__ _Float16 chp[5][32][32];   // [channel][row][col] 10240 B
    __shared__ _Float16 tmpT[16][32];     // [n][k] transpose buffer 1024 B

    const int lane = threadIdx.x;         // wave32: one wave per 16x16 tile
    const int r0 = blockIdx.y * TILE;
    const int c0 = blockIdx.x * TILE;
    const size_t ibase = (size_t)blockIdx.z * IMG_H * IMG_W;

    // ---- zero channel patches (covers pad regions) ----
    {
        float4 z4 = {0.f, 0.f, 0.f, 0.f};
        float4* p = (float4*)&chp[0][0][0];
        #pragma unroll
        for (int i = 0; i < 20; ++i)      // 640 float4 total / 32 lanes
            p[lane + 32 * i] = z4;
    }
    __syncthreads();

    // ---- load 26x26 patch (zero-padded at image edges), build 5 channels ----
    for (int i = lane; i < PATCH * PATCH; i += 32) {
        int pr = i / PATCH;
        int pc = i - pr * PATCH;
        int gr = r0 - PAD + pr;
        int gc = c0 - PAD + pc;
        bool ok = (gr >= 0) & (gr < IMG_H) & (gc >= 0) & (gc < IMG_W);
        size_t gi = ibase + (size_t)gr * IMG_W + gc;
        float a = ok ? img0[gi] : 0.f;
        float b = ok ? img1[gi] : 0.f;
        chp[0][pr][pc] = (_Float16)a;
        chp[1][pr][pc] = (_Float16)b;
        chp[2][pr][pc] = (_Float16)(a * a);
        chp[3][pr][pc] = (_Float16)(b * b);
        chp[4][pr][pc] = (_Float16)(a * b);
    }
    __syncthreads();

    const int nn   = lane & 15;   // column / M index within lane-half
    const int hi   = lane >> 4;   // lane-half select
    const int moff = hi * 8;      // C/D row offset; also A-operand K sub-base

    // ---- banded Gaussian operands (built once, reused for all 5 channels) ----
    // Horizontal pass B-operand (32x16): B[k][n] = g[k-n], 0<=k-n<=10
    v16h bH;
    {
        int kb = hi * 16;
        #pragma unroll
        for (int v = 0; v < 8; ++v) {
            int d0 = ((v >> 2) * 16) + moff + 2 * (v & 3) - nn;
            int d1 = d0 + 1;
            bH[2 * v]     = (d0 >= 0 && d0 <= 10) ? (_Float16)GK[d0] : (_Float16)0.f;
            bH[2 * v + 1] = (d1 >= 0 && d1 <= 10) ? (_Float16)GK[d1] : (_Float16)0.f;
        }
    }
    // Vertical pass A-operand (16x32): A[m][k] = g[k-m], 0<=k-m<=10
    v16h aV;
    {
        #pragma unroll
        for (int v = 0; v < 8; ++v) {
            int k0 = ((v >> 2) * 16) + moff + 2 * (v & 3);
            int d0 = k0 - nn;
            int d1 = d0 + 1;
            aV[2 * v]     = (d0 >= 0 && d0 <= 10) ? (_Float16)GK[d0] : (_Float16)0.f;
            aV[2 * v + 1] = (d1 >= 0 && d1 <= 10) ? (_Float16)GK[d1] : (_Float16)0.f;
        }
    }

    v8f acc[5];

    #pragma unroll
    for (int ch = 0; ch < 5; ++ch) {
        // ---- horizontal pass: two 16-row blocks ----
        v8f dh[2];
        #pragma unroll
        for (int b = 0; b < 2; ++b) {
            // A-operand: 16-bit 16x32 layout — lane m=nn, K striped per §7.12.2
            v16h a;
            #pragma unroll
            for (int v = 0; v < 8; ++v) {
                int k0 = ((v >> 2) * 16) + moff + 2 * (v & 3);
                h2 p = *(const h2*)&chp[ch][b * 16 + nn][k0];
                a[2 * v]     = p.x;
                a[2 * v + 1] = p.y;
            }
            v8f z = {};
            dh[b] = __builtin_amdgcn_wmma_f32_16x16x32_f16(
                false, a, false, bH, (short)0, z, false, false);
        }

        __syncthreads();  // WAR: previous channel's tmpT reads complete
        // D layout: lane holds column nn, rows m = b*16 + moff + v.
        // Store transposed as tmpT[n][k] so vertical B-operand loads are contiguous pairs.
        #pragma unroll
        for (int b = 0; b < 2; ++b) {
            #pragma unroll
            for (int v = 0; v < 8; v += 2) {
                h2 p;
                p.x = (_Float16)dh[b][v];
                p.y = (_Float16)dh[b][v + 1];
                *(h2*)&tmpT[nn][b * 16 + moff + v] = p;
            }
        }
        __syncthreads();  // RAW: transpose buffer visible

        // ---- vertical pass: B-operand (32x16) = filtered rows ----
        v16h bp;
        {
            int kb = hi * 16;
            #pragma unroll
            for (int v = 0; v < 8; ++v) {
                h2 p = *(const h2*)&tmpT[nn][((v >> 2) * 16) + moff + 2 * (v & 3)];
                bp[2 * v]     = p.x;
                bp[2 * v + 1] = p.y;
            }
        }
        v8f z = {};
        acc[ch] = __builtin_amdgcn_wmma_f32_16x16x32_f16(
            false, aV, false, bp, (short)0, z, false, false);
    }

    // ---- SSIM elementwise math + store ----
    // Vertical D layout: lane column n = nn, rows m = moff + v
    const size_t obase = ibase + (size_t)r0 * IMG_W + c0;
    #pragma unroll
    for (int v = 0; v < 8; ++v) {
        float mu0  = acc[0][v];
        float mu1  = acc[1][v];
        float mu0s = mu0 * mu0;
        float mu1s = mu1 * mu1;
        float mu01 = mu0 * mu1;
        float s00  = fmaxf(acc[2][v] - mu0s, 0.f);
        float s11  = fmaxf(acc[3][v] - mu1s, 0.f);
        float s01  = fmaxf(acc[4][v] - mu01, 0.f);
        float num  = (2.f * mu01 + SSIM_C0) * (2.f * s01 + SSIM_C1);
        float den  = (mu0s + mu1s + SSIM_C0) * (s00 + s11 + SSIM_C1);
        out[obase + (size_t)(moff + v) * IMG_W + nn] = num / den;
    }
}

extern "C" void kernel_launch(void* const* d_in, const int* in_sizes, int n_in,
                              void* d_out, int out_size, void* d_ws, size_t ws_size,
                              hipStream_t stream) {
    const float* img0 = (const float*)d_in[0];
    const float* img1 = (const float*)d_in[1];
    float* out = (float*)d_out;

    int batch = in_sizes[0] / (IMG_H * IMG_W);  // 32 for the reference shapes
    dim3 grid(IMG_W / TILE, IMG_H / TILE, batch);
    dim3 block(32);
    hipLaunchKernelGGL(ssim_wmma_kernel, grid, block, 0, stream, img0, img1, out);
}
